// Transformer_39487929319730
// MI455X (gfx1250) — compile-verified, matrix operand order checked by reading, not run on hardware
//
#include <hip/hip_runtime.h>

#define LAYERS 6
#define DMODEL 768
#define NHEAD  12
#define DHEAD  64
#define FFDIM  3072
#define SEQ    512
#define BATCH  8
#define MROWS  (BATCH * SEQ)   // 4096

typedef float  v4f   __attribute__((ext_vector_type(4)));
typedef float  v8f   __attribute__((ext_vector_type(8)));
typedef __bf16 v16bf __attribute__((ext_vector_type(16)));
typedef int    v4i   __attribute__((vector_size(16)));

union FragBF {
    v16bf  v;
    v4f    f4[2];
    __bf16 h[16];
};

#if defined(__has_builtin)
#if __has_builtin(__builtin_amdgcn_global_load_async_to_lds_b128) && \
    __has_builtin(__builtin_amdgcn_s_wait_asynccnt)
#define USE_ASYNC 1
#endif
#endif
#ifndef USE_ASYNC
#define USE_ASYNC 0
#endif

// Builtin signature (from hipcc diagnostic): param0 = v4i* in global AS,
// param1 = v4i* in LDS AS, then imm offset, imm cpol.
#define GP128(p) ((__attribute__((address_space(1))) v4i*)(p))
#define LP128(p) ((__attribute__((address_space(3))) v4i*)(p))

__device__ __forceinline__ v8f v8f_zero() {
    v8f z;
#pragma unroll
    for (int i = 0; i < 8; ++i) z[i] = 0.0f;
    return z;
}

__device__ __forceinline__ v8f wmma_bf16(const FragBF& a, const FragBF& b, v8f c) {
    // D = A(16x32 bf16) * B(32x16 bf16) + C(16x16 f32)
    return __builtin_amdgcn_wmma_f32_16x16x32_bf16(
        false, a.v, false, b.v, (short)0, c, false, false);
}

// ---------------------------------------------------------------------------
// fp32 -> bf16 (grid-stride) — used once for the input activations x
// ---------------------------------------------------------------------------
__global__ __launch_bounds__(256)
void cvt_f32_bf16(const float* __restrict__ in, __bf16* __restrict__ out, int n) {
    int i = blockIdx.x * blockDim.x + threadIdx.x;
    int stride = gridDim.x * blockDim.x;
    for (; i < n; i += stride) out[i] = (__bf16)in[i];
}

// ---------------------------------------------------------------------------
// Weight prep: W[l][k][n] fp32  ->  WT[l][n][k] bf16  (LDS-tiled transpose,
// coalesced reads and writes). Makes GEMM B tiles async-loadable slabs.
// ---------------------------------------------------------------------------
__global__ __launch_bounds__(256)
void cvt_transpose_bf16(const float* __restrict__ in, __bf16* __restrict__ out,
                        int K, int N) {
    __shared__ __bf16 t[64 * 72];  // t[k][n], stride 72
    const int l  = blockIdx.z;
    const int n0 = blockIdx.x * 64, k0 = blockIdx.y * 64;
    const float*  src = in  + (size_t)l * K * N;
    __bf16*       dst = out + (size_t)l * K * N;
    const int tid = threadIdx.x;
    const int r  = tid >> 2;         // 0..63
    const int cs = (tid & 3) * 16;   // 0,16,32,48

    const float* p = src + (size_t)(k0 + r) * N + n0 + cs;
#pragma unroll
    for (int g = 0; g < 4; ++g) {
        v4f v = *(const v4f*)(p + g * 4);
        t[r * 72 + cs + g * 4 + 0] = (__bf16)v.x;
        t[r * 72 + cs + g * 4 + 1] = (__bf16)v.y;
        t[r * 72 + cs + g * 4 + 2] = (__bf16)v.z;
        t[r * 72 + cs + g * 4 + 3] = (__bf16)v.w;
    }
    __syncthreads();
    FragBF u;
#pragma unroll
    for (int e = 0; e < 16; ++e) u.h[e] = t[(cs + e) * 72 + r];
    *(v4f*)(dst + (size_t)(n0 + r) * K + k0 + cs)     = u.f4[0];
    *(v4f*)(dst + (size_t)(n0 + r) * K + k0 + cs + 8) = u.f4[1];
}

// ---------------------------------------------------------------------------
// GEMM: out[M,N] = act(A[M,K](bf16) @ B[K,N] + bias), B given as BT[n][k] bf16.
// 256 threads = 8 wave32 waves; tile 128x64, K-step 32; LDS double-buffered,
// tiles streamed with global_load_async_to_lds_b128 (ASYNCcnt) when available.
// OUTBF=1 stores bf16 (feeds the next WMMA consumer), else fp32.
// ---------------------------------------------------------------------------
template <int GELU, int OUTBF>
__global__ __launch_bounds__(256)
void gemm_bias(const __bf16* __restrict__ A, const __bf16* __restrict__ BT,
               const float* __restrict__ bias, float* __restrict__ C,
               __bf16* __restrict__ Cb, int M, int N, int K) {
    (void)M;
    __shared__ __bf16 As[2][128 * 40];  // [m][k] stride 40
    __shared__ __bf16 Bs[2][64 * 40];   // [n][k] stride 40

    const int tid  = threadIdx.x;
    const int wv   = tid >> 5;
    const int lane = tid & 31;
    const int nl   = lane & 15;
    const int hlf  = lane >> 4;
    const int row0 = blockIdx.y * 128;
    const int col0 = blockIdx.x * 64;

    // staging assignments: A 128x32 (2 x b128 per thread), B 64x32 (1 x b128)
    const int arow = tid >> 1;
    const int aks  = (tid & 1) * 16;
    const int brow = tid >> 2;
    const int bks  = (tid & 3) * 8;

    v8f acc[4];
#pragma unroll
    for (int j = 0; j < 4; ++j) acc[j] = v8f_zero();

    auto issue = [&](int k0, int bufi) {
        const __bf16* ga = A + (size_t)(row0 + arow) * K + k0 + aks;
        __bf16*       la = &As[bufi][arow * 40 + aks];
        const __bf16* gb = BT + (size_t)(col0 + brow) * K + k0 + bks;
        __bf16*       lb = &Bs[bufi][brow * 40 + bks];
#if USE_ASYNC
        __builtin_amdgcn_global_load_async_to_lds_b128(GP128(ga), LP128(la), 0, 0);
        __builtin_amdgcn_global_load_async_to_lds_b128(GP128(ga), LP128(la), 16, 0);
        __builtin_amdgcn_global_load_async_to_lds_b128(GP128(gb), LP128(lb), 0, 0);
#else
        *(v4f*)la       = *(const v4f*)ga;
        *(v4f*)(la + 8) = *(const v4f*)(ga + 8);
        *(v4f*)lb       = *(const v4f*)gb;
#endif
    };

    const int nK = K >> 5;
    issue(0, 0);
    for (int i = 0; i < nK; ++i) {
        const int cur = i & 1;
        if (i + 1 < nK) {
            issue((i + 1) << 5, cur ^ 1);
#if USE_ASYNC
            __builtin_amdgcn_s_wait_asynccnt(3);  // tile i complete
#endif
        } else {
#if USE_ASYNC
            __builtin_amdgcn_s_wait_asynccnt(0);
#endif
        }
        __syncthreads();

        FragBF af;
        {
            const int ab = (wv * 16 + nl) * 40 + hlf * 8;
            af.f4[0] = *(const v4f*)&As[cur][ab];
            af.f4[1] = *(const v4f*)&As[cur][ab + 16];
        }
#pragma unroll
        for (int j = 0; j < 4; ++j) {
            FragBF bf;
            const int bb = (j * 16 + nl) * 40 + hlf * 8;
            bf.f4[0] = *(const v4f*)&Bs[cur][bb];
            bf.f4[1] = *(const v4f*)&Bs[cur][bb + 16];
            acc[j] = wmma_bf16(af, bf, acc[j]);
        }
        __syncthreads();
    }

#pragma unroll
    for (int j = 0; j < 4; ++j) {
        const int col = col0 + j * 16 + nl;
        const float bv = bias[col];
#pragma unroll
        for (int r = 0; r < 8; ++r) {
            const int row = row0 + wv * 16 + r + hlf * 8;
            float vv = acc[j][r] + bv;
            if (GELU) vv = 0.5f * vv * (1.0f + erff(vv * 0.70710678118654752f));
            if (OUTBF) Cb[(size_t)row * N + col] = (__bf16)vv;
            else       C[(size_t)row * N + col] = vv;
        }
    }
}

// ---------------------------------------------------------------------------
// Attention: one wave32 block per (batch, head, 16-query tile). All operands
// bf16 (produced by the QKV GEMMs); 1/sqrt(dh) folded into the f32 scores.
// ---------------------------------------------------------------------------
__global__ __launch_bounds__(32)
void attention(const __bf16* __restrict__ Q, const __bf16* __restrict__ Km,
               const __bf16* __restrict__ V, const int* __restrict__ mask,
               __bf16* __restrict__ O) {
    __shared__ float  sc[16 * 528];   // scores fp32
    __shared__ __bf16 pb[16 * 520];   // probabilities bf16
    __shared__ __bf16 vt[64 * 40];    // V^T tile: [dim][key]
    __shared__ float  inv_s[16];

    const int lane = threadIdx.x;
    const int nl   = lane & 15;
    const int hlf  = lane >> 4;
    const int qt = blockIdx.x, hh = blockIdx.y, b = blockIdx.z;

    const size_t qrow0 = (size_t)(b * SEQ + qt * 16);
    const __bf16* qp = Q + qrow0 * DMODEL + hh * DHEAD;

    FragBF qf0, qf1;  // A-operand 16x32 x2 over dh=64 — aligned b128 loads
    qf0.f4[0] = *(const v4f*)(qp + nl * DMODEL +      hlf * 8);
    qf0.f4[1] = *(const v4f*)(qp + nl * DMODEL + 16 + hlf * 8);
    qf1.f4[0] = *(const v4f*)(qp + nl * DMODEL + 32 + hlf * 8);
    qf1.f4[1] = *(const v4f*)(qp + nl * DMODEL + 48 + hlf * 8);

    // Phase 1: 16x16 score tiles across all 512 keys
    for (int kt = 0; kt < 32; ++kt) {
        const __bf16* kp = Km + (size_t)(b * SEQ + kt * 16 + nl) * DMODEL + hh * DHEAD;
        FragBF kf0, kf1;  // B-operand: lane = key col, d = 8*hlf+s, 16+8*hlf+(s-8); kf1 at +32
        kf0.f4[0] = *(const v4f*)(kp +      hlf * 8);
        kf0.f4[1] = *(const v4f*)(kp + 16 + hlf * 8);
        kf1.f4[0] = *(const v4f*)(kp + 32 + hlf * 8);
        kf1.f4[1] = *(const v4f*)(kp + 48 + hlf * 8);
        v8f s = v8f_zero();
        s = wmma_bf16(qf0, kf0, s);
        s = wmma_bf16(qf1, kf1, s);
        const int mv = mask[b * SEQ + kt * 16 + nl];
#pragma unroll
        for (int r = 0; r < 8; ++r) {
            float sv = s[r] * 0.125f;  // 1/sqrt(64)
            if (mv == 0) sv = -1e9f;
            sc[(r + hlf * 8) * 528 + kt * 16 + nl] = sv;
        }
    }
    __syncthreads();

    // Phase 2: softmax — 2 lanes per row, combine halves via shfl_xor(16)
    {
        const int row = nl, cb = hlf * 256;
        float mx = -3.0e38f;
        for (int c = 0; c < 256; ++c) mx = fmaxf(mx, sc[row * 528 + cb + c]);
        mx = fmaxf(mx, __shfl_xor(mx, 16));
        float sum = 0.0f;
        for (int c = 0; c < 256; ++c) {
            float e = __expf(sc[row * 528 + cb + c] - mx);
            pb[row * 520 + cb + c] = (__bf16)e;
            sum += e;
        }
        sum += __shfl_xor(sum, 16);
        if (hlf == 0) inv_s[row] = 1.0f / sum;
    }
    __syncthreads();

    // Phase 3: ctx = P @ V in K-steps of 32 keys
    v8f cacc[4];
#pragma unroll
    for (int j = 0; j < 4; ++j) cacc[j] = v8f_zero();

    for (int kt2 = 0; kt2 < 16; ++kt2) {
        {   // stage V^T: lane owns key row (kt2*32 + lane)
            const __bf16* vp = V + (size_t)(b * SEQ + kt2 * 32 + lane) * DMODEL + hh * DHEAD;
#pragma unroll
            for (int g = 0; g < 8; ++g) {
                v4f t = *(const v4f*)(vp + g * 8);
                const __bf16* hb = (const __bf16*)&t;
#pragma unroll
                for (int j = 0; j < 8; ++j) vt[(g * 8 + j) * 40 + lane] = hb[j];
            }
        }
        __syncthreads();
        FragBF pf;
        {
            const int pbase = nl * 520 + kt2 * 32 + hlf * 8;
            pf.f4[0] = *(const v4f*)&pb[pbase];
            pf.f4[1] = *(const v4f*)&pb[pbase + 16];
        }
#pragma unroll
        for (int j = 0; j < 4; ++j) {
            FragBF vf;
            const int vb = (j * 16 + nl) * 40 + hlf * 8;
            vf.f4[0] = *(const v4f*)&vt[vb];
            vf.f4[1] = *(const v4f*)&vt[vb + 16];
            cacc[j] = wmma_bf16(pf, vf, cacc[j]);
        }
        __syncthreads();
    }

    float iv[8];
#pragma unroll
    for (int r = 0; r < 8; ++r) iv[r] = inv_s[r + hlf * 8];
    __bf16* op = O + qrow0 * DMODEL + hh * DHEAD;
#pragma unroll
    for (int j = 0; j < 4; ++j)
#pragma unroll
        for (int r = 0; r < 8; ++r)
            op[(size_t)(r + hlf * 8) * DMODEL + j * 16 + nl] = (__bf16)(cacc[j][r] * iv[r]);
}

// ---------------------------------------------------------------------------
// LayerNorm with fused residual: out = LN(a + r)*g + beta. One wave per row.
// EMITB=1 additionally emits a bf16 twin for the downstream WMMA consumer.
// ---------------------------------------------------------------------------
template <int EMITB>
__global__ __launch_bounds__(32)
void ln_residual(const float* __restrict__ a, const float* __restrict__ r,
                 const float* __restrict__ g, const float* __restrict__ beta,
                 float* __restrict__ out, __bf16* __restrict__ outb) {
    const int row = blockIdx.x;
    const int lane = threadIdx.x;
    const size_t base = (size_t)row * DMODEL;
    float x[24];
    float sum = 0.0f, sq = 0.0f;
#pragma unroll
    for (int e = 0; e < 24; ++e) {
        const int col = e * 32 + lane;
        const float v = a[base + col] + r[base + col];
        x[e] = v;
        sum += v;
        sq += v * v;
    }
#pragma unroll
    for (int o = 16; o > 0; o >>= 1) {
        sum += __shfl_xor(sum, o);
        sq  += __shfl_xor(sq, o);
    }
    const float mu = sum * (1.0f / DMODEL);
    const float var = sq * (1.0f / DMODEL) - mu * mu;
    const float rs = rsqrtf(var + 1e-12f);
#pragma unroll
    for (int e = 0; e < 24; ++e) {
        const int col = e * 32 + lane;
        const float y = (x[e] - mu) * rs * g[col] + beta[col];
        out[base + col] = y;
        if (EMITB) outb[base + col] = (__bf16)y;
    }
}

// ---------------------------------------------------------------------------
extern "C" void kernel_launch(void* const* d_in, const int* in_sizes, int n_in,
                              void* d_out, int out_size, void* d_ws, size_t ws_size,
                              hipStream_t stream) {
    (void)in_sizes; (void)n_in; (void)out_size; (void)ws_size;

    const float* x   = (const float*)d_in[0];
    const int*   msk = (const int*)  d_in[1];
    const float* Wq  = (const float*)d_in[2];
    const float* bq  = (const float*)d_in[3];
    const float* Wk  = (const float*)d_in[4];
    const float* bk  = (const float*)d_in[5];
    const float* Wv  = (const float*)d_in[6];
    const float* bv  = (const float*)d_in[7];
    const float* Wo  = (const float*)d_in[8];
    const float* bo  = (const float*)d_in[9];
    const float* g1  = (const float*)d_in[10];
    const float* be1 = (const float*)d_in[11];
    const float* W1  = (const float*)d_in[12];
    const float* b1  = (const float*)d_in[13];
    const float* W2  = (const float*)d_in[14];
    const float* b2  = (const float*)d_in[15];
    const float* g2  = (const float*)d_in[16];
    const float* be2 = (const float*)d_in[17];

    char* ws = (char*)d_ws;
    size_t off = 0;
    auto take = [&](size_t bytes) -> char* {
        char* p = ws + off;
        off += (bytes + 255) & ~(size_t)255;
        return p;
    };

    const size_t nDD = (size_t)LAYERS * DMODEL * DMODEL;
    const size_t nDF = (size_t)LAYERS * DMODEL * FFDIM;
    __bf16* WqT = (__bf16*)take(nDD * 2);
    __bf16* WkT = (__bf16*)take(nDD * 2);
    __bf16* WvT = (__bf16*)take(nDD * 2);
    __bf16* WoT = (__bf16*)take(nDD * 2);
    __bf16* W1T = (__bf16*)take(nDF * 2);
    __bf16* W2T = (__bf16*)take(nDF * 2);

    const size_t nAct = (size_t)MROWS * DMODEL;
    __bf16* xB   = (__bf16*)take(nAct * 2);
    __bf16* fQ   = (__bf16*)take(nAct * 2);
    __bf16* fK   = (__bf16*)take(nAct * 2);
    __bf16* fV   = (__bf16*)take(nAct * 2);
    __bf16* fCtx = (__bf16*)take(nAct * 2);
    __bf16* fH2B = (__bf16*)take(nAct * 2);
    __bf16* fHB  = (__bf16*)take(nAct * 2);
    __bf16* fFF  = (__bf16*)take((size_t)MROWS * FFDIM * 2);
    float*  fSa  = (float*)take(nAct * 4);
    float*  fH2  = (float*)take(nAct * 4);
    float*  fF2  = (float*)take(nAct * 4);
    float*  fH   = (float*)take(nAct * 4);

    // weight prep: fp32 [k][n] -> bf16 [n][k]
    cvt_transpose_bf16<<<dim3(DMODEL/64, DMODEL/64, LAYERS), 256, 0, stream>>>(Wq, WqT, DMODEL, DMODEL);
    cvt_transpose_bf16<<<dim3(DMODEL/64, DMODEL/64, LAYERS), 256, 0, stream>>>(Wk, WkT, DMODEL, DMODEL);
    cvt_transpose_bf16<<<dim3(DMODEL/64, DMODEL/64, LAYERS), 256, 0, stream>>>(Wv, WvT, DMODEL, DMODEL);
    cvt_transpose_bf16<<<dim3(DMODEL/64, DMODEL/64, LAYERS), 256, 0, stream>>>(Wo, WoT, DMODEL, DMODEL);
    cvt_transpose_bf16<<<dim3(FFDIM/64,  DMODEL/64, LAYERS), 256, 0, stream>>>(W1, W1T, DMODEL, FFDIM);
    cvt_transpose_bf16<<<dim3(DMODEL/64, FFDIM/64,  LAYERS), 256, 0, stream>>>(W2, W2T, FFDIM, DMODEL);
    cvt_f32_bf16<<<2048, 256, 0, stream>>>(x, xB, (int)nAct);

    const dim3 gD(DMODEL / 64, MROWS / 128);   // (12, 32)
    const dim3 gF(FFDIM / 64, MROWS / 128);    // (48, 32)
    const dim3 gAtt(SEQ / 16, NHEAD, BATCH);   // (32, 12, 8)

    const float*  hF = x;   // fp32 residual stream
    const __bf16* hB = xB;  // bf16 WMMA stream
    for (int i = 0; i < LAYERS; ++i) {
        const size_t oDD = (size_t)i * DMODEL * DMODEL;
        const size_t oDF = (size_t)i * DMODEL * FFDIM;

        gemm_bias<0,1><<<gD, 256, 0, stream>>>(hB, WqT + oDD, bq + i*DMODEL, nullptr, fQ, MROWS, DMODEL, DMODEL);
        gemm_bias<0,1><<<gD, 256, 0, stream>>>(hB, WkT + oDD, bk + i*DMODEL, nullptr, fK, MROWS, DMODEL, DMODEL);
        gemm_bias<0,1><<<gD, 256, 0, stream>>>(hB, WvT + oDD, bv + i*DMODEL, nullptr, fV, MROWS, DMODEL, DMODEL);
        attention<<<gAtt, 32, 0, stream>>>(fQ, fK, fV, msk, fCtx);
        gemm_bias<0,0><<<gD, 256, 0, stream>>>(fCtx, WoT + oDD, bo + i*DMODEL, fSa, nullptr, MROWS, DMODEL, DMODEL);
        ln_residual<1><<<MROWS, 32, 0, stream>>>(fSa, hF, g1 + i*DMODEL, be1 + i*DMODEL, fH2, fH2B);
        gemm_bias<1,1><<<gF, 256, 0, stream>>>(fH2B, W1T + oDF, b1 + i*FFDIM, nullptr, fFF, MROWS, FFDIM, DMODEL);
        gemm_bias<0,0><<<gD, 256, 0, stream>>>(fFF, W2T + oDF, b2 + i*DMODEL, fF2, nullptr, MROWS, DMODEL, FFDIM);
        float* dst = (i == LAYERS - 1) ? (float*)d_out : fH;
        ln_residual<1><<<MROWS, 32, 0, stream>>>(fF2, fH2, g2 + i*DMODEL, be2 + i*DMODEL, dst, fHB);
        hF = fH;
        hB = fHB;
    }
}
